// Gemma3Encoder_31825707664097
// MI455X (gfx1250) — matrix operand order repaired and checked by reading, not compiled, not run on hardware
//
#include <hip/hip_runtime.h>
#include <math.h>

#define HH     768
#define NLAY   24
#define NHEAD  3
#define HD     256
#define INTER  1152
#define LSEQ   1024
#define WINSZ  512
#define NEGV   (-10000.0f)
#define EPSV   1e-6f

typedef __attribute__((ext_vector_type(16))) __bf16 v16bf;
typedef __attribute__((ext_vector_type(8)))  __bf16 v8bf;
typedef __attribute__((ext_vector_type(8)))  float  v8f;

// Native bf16 converts (v_cvt_bf16_f32 on gfx1250), RTNE
static __device__ __forceinline__ __bf16 f2bfh(float f) { return (__bf16)f; }
static __device__ __forceinline__ unsigned short f2bfu(float f) {
  return __builtin_bit_cast(unsigned short, (__bf16)f);
}

// ---------------------------------------------------------------------------
// Embedding gather + sqrt(H) scale
// ---------------------------------------------------------------------------
__global__ __launch_bounds__(256) void k_embed(const int* __restrict__ ids,
                                               const float* __restrict__ emb,
                                               float* __restrict__ h) {
  int pos = blockIdx.x;
  const float* row = emb + (size_t)ids[pos] * HH;
  float* o = h + (size_t)pos * HH;
  for (int i = threadIdx.x; i < HH; i += 256) o[i] = row[i] * 27.712812921102035f;
}

// ---------------------------------------------------------------------------
// Gemma RMSNorm -> bf16 activations (WMMA A operand)
// ---------------------------------------------------------------------------
__global__ __launch_bounds__(256) void k_rmsnorm_bf16(const float* __restrict__ x,
                                                      const float* __restrict__ w,
                                                      unsigned short* __restrict__ out,
                                                      int width) {
  int row = blockIdx.x;
  const float* xr = x + (size_t)row * width;
  unsigned short* orow = out + (size_t)row * width;
  __shared__ float red[256];
  float s = 0.f;
  for (int i = threadIdx.x; i < width; i += 256) { float v = xr[i]; s += v * v; }
  red[threadIdx.x] = s; __syncthreads();
  for (int off = 128; off > 0; off >>= 1) {
    if (threadIdx.x < off) red[threadIdx.x] += red[threadIdx.x + off];
    __syncthreads();
  }
  float scale = rsqrtf(red[0] / (float)width + EPSV);
  for (int i = threadIdx.x; i < width; i += 256)
    orow[i] = f2bfu(xr[i] * scale * (1.f + w[i]));
}

// h += rms(x, w)
__global__ __launch_bounds__(256) void k_resid_rms_add(float* __restrict__ h,
                                                       const float* __restrict__ x,
                                                       const float* __restrict__ w,
                                                       int width) {
  int row = blockIdx.x;
  const float* xr = x + (size_t)row * width;
  float* hr = h + (size_t)row * width;
  __shared__ float red[256];
  float s = 0.f;
  for (int i = threadIdx.x; i < width; i += 256) { float v = xr[i]; s += v * v; }
  red[threadIdx.x] = s; __syncthreads();
  for (int off = 128; off > 0; off >>= 1) {
    if (threadIdx.x < off) red[threadIdx.x] += red[threadIdx.x + off];
    __syncthreads();
  }
  float scale = rsqrtf(red[0] / (float)width + EPSV);
  for (int i = threadIdx.x; i < width; i += 256)
    hr[i] += xr[i] * scale * (1.f + w[i]);
}

// final norm -> fp32 output
__global__ __launch_bounds__(256) void k_rms_out(const float* __restrict__ x,
                                                 const float* __restrict__ w,
                                                 float* __restrict__ out) {
  int row = blockIdx.x;
  const float* xr = x + (size_t)row * HH;
  float* orow = out + (size_t)row * HH;
  __shared__ float red[256];
  float s = 0.f;
  for (int i = threadIdx.x; i < HH; i += 256) { float v = xr[i]; s += v * v; }
  red[threadIdx.x] = s; __syncthreads();
  for (int off = 128; off > 0; off >>= 1) {
    if (threadIdx.x < off) red[threadIdx.x] += red[threadIdx.x + off];
    __syncthreads();
  }
  float scale = rsqrtf(red[0] / (float)HH + EPSV);
  for (int i = threadIdx.x; i < HH; i += 256)
    orow[i] = xr[i] * scale * (1.f + w[i]);
}

// ---------------------------------------------------------------------------
// Per-head RMSNorm over HD + RoPE -> bf16 [head, pos, HD]
// blockDim = HD (=256); grid = (LSEQ, nheads)
// ---------------------------------------------------------------------------
__global__ __launch_bounds__(256) void k_rms_rope(const float* __restrict__ x,
                                                  const float* __restrict__ w,
                                                  unsigned short* __restrict__ out,
                                                  int nh, float logTheta) {
  int pos = blockIdx.x, head = blockIdx.y, d = threadIdx.x;
  const float* xr = x + (size_t)pos * ((size_t)nh * HD) + (size_t)head * HD;
  __shared__ float xs[HD];
  __shared__ float red[HD];
  float v = xr[d];
  xs[d] = v;
  red[d] = v * v;
  __syncthreads();
  for (int off = 128; off > 0; off >>= 1) {
    if (d < off) red[d] += red[d + off];
    __syncthreads();
  }
  float scale = rsqrtf(red[0] * (1.f / (float)HD) + EPSV);
  int fi = d & 127;
  float ang = (float)pos * __expf(-logTheta * (float)fi * (1.f / 128.f));
  float nv = v * scale * (1.f + w[d]);
  int d2 = (d < 128) ? d + 128 : d - 128;
  float pv = xs[d2] * scale * (1.f + w[d2]);
  float rot = (d < 128) ? -pv : pv;
  out[((size_t)head * LSEQ + pos) * HD + d] = f2bfu(nv * __cosf(ang) + rot * __sinf(ang));
}

// v [L, HD] fp32 -> vt [HD, L] bf16
__global__ __launch_bounds__(256) void k_v_t(const float* __restrict__ v,
                                             unsigned short* __restrict__ vt) {
  int pos = blockIdx.x, d = threadIdx.x;
  vt[(size_t)d * LSEQ + pos] = f2bfu(v[(size_t)pos * HD + d]);
}

__global__ void k_cvt_bf16(const float* __restrict__ in, unsigned short* __restrict__ out, int n) {
  int i = blockIdx.x * blockDim.x + threadIdx.x;
  if (i < n) out[i] = f2bfu(in[i]);
}

__global__ void k_gelu_mul(const float* __restrict__ g, const float* __restrict__ u,
                           unsigned short* __restrict__ out, int n) {
  int i = blockIdx.x * blockDim.x + threadIdx.x;
  if (i >= n) return;
  float x = g[i];
  float t = 0.5f * x * (1.f + tanhf(0.7978845608028654f * (x + 0.044715f * x * x * x)));
  out[i] = f2bfu(t * u[i]);
}

// ---------------------------------------------------------------------------
// Mask + softmax per (head, query row); bf16 probs out
// ---------------------------------------------------------------------------
__global__ __launch_bounds__(256) void k_softmax(const float* __restrict__ scores,
                                                 const float* __restrict__ amask,
                                                 unsigned short* __restrict__ probs,
                                                 int isFull) {
  int qi = blockIdx.x, hh = blockIdx.y, t = threadIdx.x;
  const float* srow = scores + ((size_t)hh * LSEQ + qi) * LSEQ;
  unsigned short* prow = probs + ((size_t)hh * LSEQ + qi) * LSEQ;
  __shared__ float red[256];
  float vals[4];
  float mx = -3.0e38f;
#pragma unroll
  for (int j = 0; j < 4; ++j) {
    int kj = t + j * 256;
    float m = (1.f - amask[kj]) * NEGV;
    if (!isFull) {
      int dist = qi > kj ? qi - kj : kj - qi;
      if (dist > WINSZ) m += NEGV;
    }
    float s = srow[kj] + m;
    vals[j] = s;
    mx = fmaxf(mx, s);
  }
  red[t] = mx; __syncthreads();
  for (int off = 128; off > 0; off >>= 1) {
    if (t < off) red[t] = fmaxf(red[t], red[t + off]);
    __syncthreads();
  }
  mx = red[0]; __syncthreads();
  float sum = 0.f;
#pragma unroll
  for (int j = 0; j < 4; ++j) { vals[j] = __expf(vals[j] - mx); sum += vals[j]; }
  red[t] = sum; __syncthreads();
  for (int off = 128; off > 0; off >>= 1) {
    if (t < off) red[t] += red[t + off];
    __syncthreads();
  }
  float inv = 1.f / red[0];
#pragma unroll
  for (int j = 0; j < 4; ++j) prow[t + j * 256] = f2bfu(vals[j] * inv);
}

// ---------------------------------------------------------------------------
// WMMA GEMM:  C[M,N] f32 = A[M,K] bf16 x W[N,K]^T
// One wave -> 16(M) x 64(N): 4 accumulators share one A fragment per K step.
// 8 waves/block -> 128 M rows. grid = (N/64, M/128, batch).
//
// A-fragment (16x32 bf16): lane = 16*half + row; elements K = half*16 + e
//                          (two 16B loads)
// B-fragment (32x16):      lane: col = lane&15, elements K = half*16 + e
// C/D:                     VGPR r, lane -> (M = r + 8*half, N = lane&15)
// ---------------------------------------------------------------------------

// fp32 weights, in-register cvt to bf16 (weight streaming is the roofline)
__global__ __launch_bounds__(256) void k_gemm_wf32(const unsigned short* __restrict__ Au,
                                                   const float* __restrict__ W,
                                                   float* __restrict__ C,
                                                   int K, int lda, int ldw, int ldc,
                                                   long long sA, long long sW, long long sC) {
  const __bf16* A = reinterpret_cast<const __bf16*>(Au) + (size_t)blockIdx.z * sA;
  W += (size_t)blockIdx.z * sW;
  C += (size_t)blockIdx.z * sC;
  int lane = threadIdx.x & 31;
  int wave = threadIdx.x >> 5;
  int m0 = (blockIdx.y * 8 + wave) * 16;
  int n0 = blockIdx.x * 64;
  int col = lane & 15, half = lane >> 4;
  const __bf16* arow = A + (size_t)(m0 + col) * lda;
  const float* wbase = W + (size_t)(n0 + col) * ldw + half * 16;
  v8f acc[4] = {{}, {}, {}, {}};
  for (int k0 = 0; k0 < K; k0 += 32) {
    v8bf a0 = *reinterpret_cast<const v8bf*>(arow + k0 + half * 16);
    v8bf a1 = *reinterpret_cast<const v8bf*>(arow + k0 + half * 16 + 8);
    v16bf a = __builtin_shufflevector(a0, a1, 0, 1, 2, 3, 4, 5, 6, 7,
                                      8, 9, 10, 11, 12, 13, 14, 15);
#pragma unroll
    for (int j = 0; j < 4; ++j) {
      const float* wp = wbase + (size_t)j * 16 * ldw + k0;
      __builtin_prefetch(wp + 32, 0, 1);      // global_prefetch next K tile
      v16bf b;
#pragma unroll
      for (int e = 0; e < 16; ++e) b[e] = f2bfh(wp[e]);
      acc[j] = __builtin_amdgcn_wmma_f32_16x16x32_bf16(false, a, false, b,
                                                       (short)0, acc[j], false, false);
    }
  }
#pragma unroll
  for (int j = 0; j < 4; ++j) {
    float* cj = C + n0 + j * 16 + col;
#pragma unroll
    for (int r = 0; r < 8; ++r)
      cj[(size_t)(m0 + r + 8 * half) * ldc] = acc[j][r];
  }
}

// Both operands bf16 (QK^T, probs@V). The 64x32 B tile is shared by all 8
// waves -> stage it in LDS once per K step with CDNA5 async global->LDS copy
// (ASYNCcnt), then read fragments with ds loads.
__global__ __launch_bounds__(256) void k_gemm_wbf16(const unsigned short* __restrict__ Au,
                                                    const unsigned short* __restrict__ Wu,
                                                    float* __restrict__ C,
                                                    int K, int lda, int ldw, int ldc,
                                                    long long sA, long long sW, long long sC) {
  const __bf16* A = reinterpret_cast<const __bf16*>(Au) + (size_t)blockIdx.z * sA;
  const __bf16* W = reinterpret_cast<const __bf16*>(Wu) + (size_t)blockIdx.z * sW;
  C += (size_t)blockIdx.z * sC;
  int t = threadIdx.x;
  int lane = t & 31;
  int wave = t >> 5;
  int m0 = (blockIdx.y * 8 + wave) * 16;
  int n0 = blockIdx.x * 64;
  int col = lane & 15, half = lane >> 4;
  const __bf16* arow = A + (size_t)(m0 + col) * lda;

  // B staging: 64 rows (N) x 32 (K) bf16 = 4KB. thread t copies 16B:
  // row = t>>2, 8-element chunk = t&3.
  __shared__ __align__(16) __bf16 smemB[64 * 32];
  const __bf16* gsrc0 = W + (size_t)(n0 + (t >> 2)) * ldw + (t & 3) * 8;
  unsigned ldsDst = (unsigned)(size_t)(&smemB[t * 8]);   // LDS byte offset (low 32 bits)

  v8f acc[4] = {{}, {}, {}, {}};
  for (int k0 = 0; k0 < K; k0 += 32) {
    // async copy this K-tile of B into LDS (one b128 per lane)
    const __bf16* gsrc = gsrc0 + k0;
    asm volatile("global_load_async_to_lds_b128 %0, %1, off"
                 :: "v"(ldsDst), "v"(gsrc) : "memory");
    asm volatile("s_wait_asynccnt 0x0" ::: "memory");
    __syncthreads();

    v8bf a0 = *reinterpret_cast<const v8bf*>(arow + k0 + half * 16);
    v8bf a1 = *reinterpret_cast<const v8bf*>(arow + k0 + half * 16 + 8);
    v16bf a = __builtin_shufflevector(a0, a1, 0, 1, 2, 3, 4, 5, 6, 7,
                                      8, 9, 10, 11, 12, 13, 14, 15);
#pragma unroll
    for (int j = 0; j < 4; ++j) {
      const __bf16* bp = &smemB[(j * 16 + col) * 32 + half * 16];
      v8bf b0 = *reinterpret_cast<const v8bf*>(bp);
      v8bf b1 = *reinterpret_cast<const v8bf*>(bp + 8);
      v16bf b = __builtin_shufflevector(b0, b1, 0, 1, 2, 3, 4, 5, 6, 7,
                                        8, 9, 10, 11, 12, 13, 14, 15);
      acc[j] = __builtin_amdgcn_wmma_f32_16x16x32_bf16(false, a, false, b,
                                                       (short)0, acc[j], false, false);
    }
    __syncthreads();   // protect smemB before next K step overwrites it
  }
#pragma unroll
  for (int j = 0; j < 4; ++j) {
    float* cj = C + n0 + j * 16 + col;
#pragma unroll
    for (int r = 0; r < 8; ++r)
      cj[(size_t)(m0 + r + 8 * half) * ldc] = acc[j][r];
  }
}

// ---------------------------------------------------------------------------
// Host orchestration
// ---------------------------------------------------------------------------
extern "C" void kernel_launch(void* const* d_in, const int* in_sizes, int n_in,
                              void* d_out, int out_size, void* d_ws, size_t ws_size,
                              hipStream_t stream) {
  (void)in_sizes; (void)n_in; (void)out_size; (void)ws_size;

  const int*   ids   = (const int*)d_in[0];
  const float* amask = (const float*)d_in[1];
  const float* emb   = (const float*)d_in[2];
  const float* Wq    = (const float*)d_in[3];
  const float* Wk    = (const float*)d_in[4];
  const float* Wv    = (const float*)d_in[5];
  const float* Wo    = (const float*)d_in[6];
  const float* qn    = (const float*)d_in[7];
  const float* kn    = (const float*)d_in[8];
  const float* ln_in = (const float*)d_in[9];
  const float* ln_pa = (const float*)d_in[10];
  const float* ln_pf = (const float*)d_in[11];
  const float* ln_pff= (const float*)d_in[12];
  const float* Wg    = (const float*)d_in[13];
  const float* Wu    = (const float*)d_in[14];
  const float* Wd    = (const float*)d_in[15];
  const float* fnw   = (const float*)d_in[16];
  float* out = (float*)d_out;

  char* base = (char*)d_ws;
  size_t off = 0;
  auto alloc = [&](size_t bytes) -> char* {
    char* p = base + off;
    off += (bytes + 255) & ~(size_t)255;
    return p;
  };

  float*          h      = (float*)         alloc((size_t)LSEQ * HH * 4);
  unsigned short* nbf    = (unsigned short*)alloc((size_t)LSEQ * INTER * 2);
  float*          qb     = (float*)         alloc((size_t)LSEQ * NHEAD * HD * 4);
  float*          kb     = (float*)         alloc((size_t)LSEQ * HD * 4);
  float*          vb     = (float*)         alloc((size_t)LSEQ * HD * 4);
  unsigned short* qbf    = (unsigned short*)alloc((size_t)NHEAD * LSEQ * HD * 2);
  unsigned short* kbf    = (unsigned short*)alloc((size_t)LSEQ * HD * 2);
  unsigned short* vtb    = (unsigned short*)alloc((size_t)HD * LSEQ * 2);
  float*          scores = (float*)         alloc((size_t)NHEAD * LSEQ * LSEQ * 4);
  unsigned short* probs  = (unsigned short*)alloc((size_t)NHEAD * LSEQ * LSEQ * 2);
  float*          attn_o = (float*)         alloc((size_t)LSEQ * HH * 4);
  unsigned short* obf    = (unsigned short*)alloc((size_t)LSEQ * HH * 2);
  float*          oproj  = (float*)         alloc((size_t)LSEQ * HH * 4);
  float*          gate   = (float*)         alloc((size_t)LSEQ * INTER * 4);
  float*          up     = (float*)         alloc((size_t)LSEQ * INTER * 4);
  unsigned short* mbf    = (unsigned short*)alloc((size_t)LSEQ * INTER * 2);
  float*          ffn    = (float*)         alloc((size_t)LSEQ * HH * 4);

  k_embed<<<LSEQ, 256, 0, stream>>>(ids, emb, h);

  for (int l = 0; l < NLAY; ++l) {
    int isFull = ((l + 1) % 6 == 0) ? 1 : 0;
    float logTheta = logf(isFull ? 1000000.0f : 10000.0f);
    const float* Wq_l = Wq + (size_t)l * NHEAD * HD * HH;
    const float* Wk_l = Wk + (size_t)l * HD * HH;
    const float* Wv_l = Wv + (size_t)l * HD * HH;
    const float* Wo_l = Wo + (size_t)l * HH * (NHEAD * HD);
    const float* Wg_l = Wg + (size_t)l * INTER * HH;
    const float* Wu_l = Wu + (size_t)l * INTER * HH;
    const float* Wd_l = Wd + (size_t)l * HH * INTER;

    // ---- attention block ----
    k_rmsnorm_bf16<<<LSEQ, 256, 0, stream>>>(h, ln_in + (size_t)l * HH, nbf, HH);

    k_gemm_wf32<<<dim3((NHEAD * HD) / 64, LSEQ / 128, 1), 256, 0, stream>>>(
        nbf, Wq_l, qb, HH, HH, HH, NHEAD * HD, 0, 0, 0);
    k_gemm_wf32<<<dim3(HD / 64, LSEQ / 128, 1), 256, 0, stream>>>(
        nbf, Wk_l, kb, HH, HH, HH, HD, 0, 0, 0);
    k_gemm_wf32<<<dim3(HD / 64, LSEQ / 128, 1), 256, 0, stream>>>(
        nbf, Wv_l, vb, HH, HH, HH, HD, 0, 0, 0);

    k_rms_rope<<<dim3(LSEQ, NHEAD), HD, 0, stream>>>(qb, qn + (size_t)l * HD, qbf, NHEAD, logTheta);
    k_rms_rope<<<dim3(LSEQ, 1),     HD, 0, stream>>>(kb, kn + (size_t)l * HD, kbf, 1, logTheta);
    k_v_t<<<LSEQ, HD, 0, stream>>>(vb, vtb);

    // scores[h] = q[h] @ k^T  (K shared across heads: strideW = 0)
    k_gemm_wbf16<<<dim3(LSEQ / 64, LSEQ / 128, NHEAD), 256, 0, stream>>>(
        qbf, kbf, scores, HD, HD, HD, LSEQ,
        (long long)LSEQ * HD, 0, (long long)LSEQ * LSEQ);

    k_softmax<<<dim3(LSEQ, NHEAD), 256, 0, stream>>>(scores, amask, probs, isFull);

    // o[h] = probs[h] @ v -> concat [L, NH*HD] via ldc=768 and head column offset
    k_gemm_wbf16<<<dim3(HD / 64, LSEQ / 128, NHEAD), 256, 0, stream>>>(
        probs, vtb, attn_o, LSEQ, LSEQ, LSEQ, HH,
        (long long)LSEQ * LSEQ, 0, (long long)HD);

    k_cvt_bf16<<<(LSEQ * HH + 255) / 256, 256, 0, stream>>>(attn_o, obf, LSEQ * HH);
    k_gemm_wf32<<<dim3(HH / 64, LSEQ / 128, 1), 256, 0, stream>>>(
        obf, Wo_l, oproj, HH, HH, HH, HH, 0, 0, 0);

    k_resid_rms_add<<<LSEQ, 256, 0, stream>>>(h, oproj, ln_pa + (size_t)l * HH, HH);

    // ---- MLP block ----
    k_rmsnorm_bf16<<<LSEQ, 256, 0, stream>>>(h, ln_pf + (size_t)l * HH, nbf, HH);

    k_gemm_wf32<<<dim3(INTER / 64, LSEQ / 128, 1), 256, 0, stream>>>(
        nbf, Wg_l, gate, HH, HH, HH, INTER, 0, 0, 0);
    k_gemm_wf32<<<dim3(INTER / 64, LSEQ / 128, 1), 256, 0, stream>>>(
        nbf, Wu_l, up, HH, HH, HH, INTER, 0, 0, 0);

    k_gelu_mul<<<(LSEQ * INTER + 255) / 256, 256, 0, stream>>>(gate, up, mbf, LSEQ * INTER);

    k_gemm_wf32<<<dim3(HH / 64, LSEQ / 128, 1), 256, 0, stream>>>(
        mbf, Wd_l, ffn, INTER, INTER, INTER, HH, 0, 0, 0);

    k_resid_rms_add<<<LSEQ, 256, 0, stream>>>(h, ffn, ln_pff + (size_t)l * HH, HH);
  }

  k_rms_out<<<LSEQ, 256, 0, stream>>>(h, fnw, out);
}
